// TFGroupViTAssignAttention_55336358643105
// MI455X (gfx1250) — matrix operand order repaired and checked by reading, not compiled, not run on hardware
//
#include <hip/hip_runtime.h>
#include <hip/hip_bf16.h>
#include <stdint.h>

// Problem constants: B=32, G=64, S=4096, D=768
#define BB 32
#define GG 64
#define SS 4096
#define DD 768

typedef __bf16 bf16;
typedef __attribute__((ext_vector_type(16))) __bf16 v16bf;
typedef __attribute__((ext_vector_type(8)))  float  v8f;

#define LANE ((int)(threadIdx.x & 31))
#define WAVE ((int)(threadIdx.x >> 5))

__device__ __forceinline__ v8f wmma_bf16(v16bf a, v16bf b, v8f c) {
  // v_wmma_f32_16x16x32_bf16: D = A(16x32) * B(32x16) + C(16x16 f32)
  return __builtin_amdgcn_wmma_f32_16x16x32_bf16(
      /*neg_a=*/false, a, /*neg_b=*/false, b,
      /*c_mod=*/(short)0, c, /*reuse_a=*/false, /*reuse_b=*/false);
}

// ---- A fragment (16x32 MxK), A row-major with leading dim lda.
// lanes 0-15: row=lane, K={0..7,16..23}; lanes 16-31: row=lane-16, K={8..15,24..31}
__device__ __forceinline__ v16bf load_a_bf16(const bf16* __restrict__ A, int lda,
                                             int row0, int k0) {
  int lane = LANE; int half = (lane >> 4) & 1; int r = lane & 15;
  const bf16* base = A + (size_t)(row0 + r) * lda + k0 + 8 * half;
  v16bf a;
#pragma unroll
  for (int v = 0; v < 8; ++v) {
    int kp = (v < 4) ? 2 * v : 8 + 2 * v;   // 0,2,4,6,16,18,20,22
    a[2 * v]     = base[kp];
    a[2 * v + 1] = base[kp + 1];
  }
  return a;
}

// Same A fragment but fp32 source, converted to bf16 on load.
// Non-temporal: the big streaming operands (key) don't fit L2; don't pollute it.
__device__ __forceinline__ v16bf load_a_f32_nt(const float* __restrict__ A, int lda,
                                               int row0, int k0) {
  int lane = LANE; int half = (lane >> 4) & 1; int r = lane & 15;
  const float* base = A + (size_t)(row0 + r) * lda + k0 + 8 * half;
  v16bf a;
#pragma unroll
  for (int v = 0; v < 8; ++v) {
    int kp = (v < 4) ? 2 * v : 8 + 2 * v;
    a[2 * v]     = (bf16)__builtin_nontemporal_load(base + kp);
    a[2 * v + 1] = (bf16)__builtin_nontemporal_load(base + kp + 1);
  }
  return a;
}

// ---- B fragment (32x16 KxN) from a TRANSPOSED store: element (k,n) at B[n*ld + k].
// Per-lane two contiguous 16-byte chunks -> 2x global_load_b128.
// lanes 0-15: n=lane, K={0..7,16..23}; lanes 16-31: n=lane-16, K={8..15,24..31}.
__device__ __forceinline__ v16bf load_bt_bf16(const bf16* __restrict__ Bm, size_t ld,
                                              int k0, int col0) {
  int lane = LANE; int half = (lane >> 4) & 1; int n = lane & 15;
  const bf16* base = Bm + (size_t)(col0 + n) * ld + k0 + 8 * half;
  v16bf b;
#pragma unroll
  for (int v = 0; v < 16; ++v) b[v] = base[v < 8 ? v : v + 8];
  return b;
}

// ================= kernel 0: weights fp32 -> TRANSPOSED bf16 ================
// dst layout: Wt[n*DD + k] = W[k*DD + n] for each of the 4 matrices.
__global__ __launch_bounds__(256) void cvt_w_kernel(const float* __restrict__ qw,
                                                    const float* __restrict__ kw,
                                                    const float* __restrict__ vw,
                                                    const float* __restrict__ pw,
                                                    bf16* __restrict__ dst) {
  const size_t WE = (size_t)DD * DD;  // 589824
  size_t i = (size_t)blockIdx.x * 256 + threadIdx.x;
  if (i < WE) {
    int n = (int)(i / DD), k = (int)(i % DD);
    size_t src = (size_t)k * DD + n;
    dst[i]          = (bf16)qw[src];
    dst[WE + i]     = (bf16)kw[src];
    dst[2 * WE + i] = (bf16)vw[src];
    dst[3 * WE + i] = (bf16)pw[src];
  }
}

// ================= generic WMMA GEMM: Out = A @ W + bias ====================
// Wt is the transposed weight [N,K] row-major. Block = 384 threads = 12 waves;
// one block covers the whole N=768 (12 waves x 64 cols) so A is read ONCE.
// Each wave computes a 32x64 tile (2 M x 4 N accumulators). grid = (M/32, 1).
// K-loop unrolled by 64 with ping-pong buffers (no register copies).
// OTRANS: store output transposed, Out[(col)*ldo + row] (used for v_t).
template <bool AF32, bool OTRANS, bool OF32>
__global__ __launch_bounds__(384) void gemm_kernel(const void* __restrict__ Araw,
                                                   const bf16* __restrict__ Wt,
                                                   const float* __restrict__ bias,
                                                   void* __restrict__ Oraw,
                                                   int M, int K, int N, size_t ldo) {
  int row0 = blockIdx.x * 32;
  int col0 = WAVE * 64;

  v8f c[2][4];
#pragma unroll
  for (int i = 0; i < 2; ++i)
#pragma unroll
    for (int j = 0; j < 4; ++j) c[i][j] = (v8f){0.f,0.f,0.f,0.f,0.f,0.f,0.f,0.f};

  v16bf a0[2], b0[4], a1[2], b1[4];
#pragma unroll
  for (int i = 0; i < 2; ++i) {
    if constexpr (AF32) a0[i] = load_a_f32_nt((const float*)Araw, K, row0 + 16 * i, 0);
    else                a0[i] = load_a_bf16((const bf16*)Araw, K, row0 + 16 * i, 0);
  }
#pragma unroll
  for (int j = 0; j < 4; ++j) b0[j] = load_bt_bf16(Wt, K, 0, col0 + 16 * j);

  for (int k0 = 0; k0 < K; k0 += 64) {  // K is a multiple of 64
    // fill buffer 1 for k0+32 (always valid), then compute buffer 0
#pragma unroll
    for (int i = 0; i < 2; ++i) {
      if constexpr (AF32) a1[i] = load_a_f32_nt((const float*)Araw, K, row0 + 16 * i, k0 + 32);
      else                a1[i] = load_a_bf16((const bf16*)Araw, K, row0 + 16 * i, k0 + 32);
    }
#pragma unroll
    for (int j = 0; j < 4; ++j) b1[j] = load_bt_bf16(Wt, K, k0 + 32, col0 + 16 * j);
#pragma unroll
    for (int i = 0; i < 2; ++i)
#pragma unroll
      for (int j = 0; j < 4; ++j) c[i][j] = wmma_bf16(a0[i], b0[j], c[i][j]);

    // fill buffer 0 for k0+64 (if any), then compute buffer 1
    if (k0 + 64 < K) {
#pragma unroll
      for (int i = 0; i < 2; ++i) {
        if constexpr (AF32) a0[i] = load_a_f32_nt((const float*)Araw, K, row0 + 16 * i, k0 + 64);
        else                a0[i] = load_a_bf16((const bf16*)Araw, K, row0 + 16 * i, k0 + 64);
      }
#pragma unroll
      for (int j = 0; j < 4; ++j) b0[j] = load_bt_bf16(Wt, K, k0 + 64, col0 + 16 * j);
    }
#pragma unroll
    for (int i = 0; i < 2; ++i)
#pragma unroll
      for (int j = 0; j < 4; ++j) c[i][j] = wmma_bf16(a1[i], b1[j], c[i][j]);
  }

  int lane = LANE; int half = (lane >> 4) & 1; int n = lane & 15;
#pragma unroll
  for (int i = 0; i < 2; ++i)
#pragma unroll
    for (int j = 0; j < 4; ++j) {
      int colg = col0 + 16 * j + n;
      float bv = bias[colg];
      if constexpr (OTRANS) {
        // rows row0+16i+8*half .. +7 contiguous -> one 16B store
        union { bf16 h[8]; uint4 u; } pk;
#pragma unroll
        for (int r = 0; r < 8; ++r) pk.h[r] = (bf16)(c[i][j][r] + bv);
        *(uint4*)((bf16*)Oraw + (size_t)colg * ldo + row0 + 16 * i + 8 * half) = pk.u;
      } else {
#pragma unroll
        for (int r = 0; r < 8; ++r) {
          int m = 16 * i + r + 8 * half;
          float val = c[i][j][r] + bv;
          if constexpr (OF32) ((float*)Oraw)[(size_t)(row0 + m) * ldo + colg] = val;
          else                ((bf16*)Oraw)[(size_t)(row0 + m) * ldo + colg] = (bf16)val;
        }
      }
    }
}

// ======== kernel: logits (q.kT) + group-axis softmax + argmax/counts ========
// grid = (S/128, B); block 256 = 8 waves. Workgroup covers all G=64 rows x 128 tokens.
__global__ __launch_bounds__(256) void logits_softmax_kernel(const bf16* __restrict__ qb,
                                                             const bf16* __restrict__ kb,
                                                             float* __restrict__ y_soft,
                                                             int* __restrict__ idx,
                                                             unsigned* __restrict__ cnt) {
  __shared__ float lds[GG][132];
  int b = blockIdx.y;
  int s0 = blockIdx.x * 128;
  int tok0 = s0 + WAVE * 16;
  const bf16* qbase = qb + (size_t)b * GG * DD;
  const bf16* kbase = kb + (size_t)b * SS * DD;

  v8f acc[4];
#pragma unroll
  for (int mt = 0; mt < 4; ++mt) acc[mt] = (v8f){0.f,0.f,0.f,0.f,0.f,0.f,0.f,0.f};

  // k rows (tokens) stored [S,D] row-major => per-lane contiguous B fragment
  v16bf bf0 = load_bt_bf16(kbase, DD, 0, tok0), bf1;
  v16bf af0[4], af1[4];
#pragma unroll
  for (int mt = 0; mt < 4; ++mt) af0[mt] = load_a_bf16(qbase, DD, mt * 16, 0);

  for (int k0 = 0; k0 < DD; k0 += 64) {   // DD multiple of 64
    bf1 = load_bt_bf16(kbase, DD, k0 + 32, tok0);
#pragma unroll
    for (int mt = 0; mt < 4; ++mt) af1[mt] = load_a_bf16(qbase, DD, mt * 16, k0 + 32);
#pragma unroll
    for (int mt = 0; mt < 4; ++mt) acc[mt] = wmma_bf16(af0[mt], bf0, acc[mt]);
    if (k0 + 64 < DD) {
      bf0 = load_bt_bf16(kbase, DD, k0 + 64, tok0);
#pragma unroll
      for (int mt = 0; mt < 4; ++mt) af0[mt] = load_a_bf16(qbase, DD, mt * 16, k0 + 64);
    }
#pragma unroll
    for (int mt = 0; mt < 4; ++mt) acc[mt] = wmma_bf16(af1[mt], bf1, acc[mt]);
  }

  const float scale = 0.03608439182435161f;  // 768^-0.5
  int lane = LANE; int half = (lane >> 4) & 1; int n = lane & 15;
#pragma unroll
  for (int mt = 0; mt < 4; ++mt)
#pragma unroll
    for (int r = 0; r < 8; ++r)
      lds[mt * 16 + r + 8 * half][WAVE * 16 + n] = acc[mt][r] * scale;
  __syncthreads();

  int ccol = threadIdx.x;  // one thread per token column
  if (ccol < 128) {
    float mx = -3.4e38f; int am = 0;
    for (int g = 0; g < GG; ++g) {
      float t = lds[g][ccol];
      if (t > mx) { mx = t; am = g; }   // first-max, matches jnp.argmax
    }
    float sum = 0.f;
    for (int g = 0; g < GG; ++g) {
      float e = __expf(lds[g][ccol] - mx);
      sum += e;
      lds[g][ccol] = e;
    }
    float inv = 1.0f / sum;
    int s = s0 + ccol;
    for (int g = 0; g < GG; ++g)
      y_soft[((size_t)b * GG + g) * SS + s] = lds[g][ccol] * inv;
    idx[b * SS + s] = am;
    atomicAdd(&cnt[b * GG + am], 1u);
  }
}

// ======== kernel: out_pre = normalize(one_hot(idx)) @ v  (WMMA) =============
// v stored transposed: v_t[d][b*S + s], ld = B*S -> contiguous B fragments.
// grid = (D/128, B); block 256 = 8 waves; wave owns 16 d-columns, all 4 G tiles.
__global__ __launch_bounds__(256) void assign_v_kernel(const int* __restrict__ idx,
                                                       const unsigned* __restrict__ cnt,
                                                       const bf16* __restrict__ v_t,
                                                       bf16* __restrict__ out_pre) {
  __shared__ int sidx[SS];
  int b = blockIdx.y;
  int d0 = blockIdx.x * 128 + WAVE * 16;
  for (int i = threadIdx.x; i < SS; i += 256) sidx[i] = idx[b * SS + i];
  __syncthreads();

  const bf16* vbase = v_t + (size_t)b * SS;   // (k=s, n=d) at vbase[d*(B*S) + s]
  const size_t ldv = (size_t)BB * SS;

  v8f acc[4];
#pragma unroll
  for (int mt = 0; mt < 4; ++mt) acc[mt] = (v8f){0.f,0.f,0.f,0.f,0.f,0.f,0.f,0.f};

  int lane = LANE; int half = (lane >> 4) & 1; int rr = lane & 15;
  const bf16 one = (bf16)1.0f, zero = (bf16)0.0f;

  v16bf bf0 = load_bt_bf16(vbase, ldv, 0, d0), bf1;
  for (int k0 = 0; k0 < SS; k0 += 64) {   // SS multiple of 64
    bf1 = load_bt_bf16(vbase, ldv, k0 + 32, d0);
#pragma unroll
    for (int mt = 0; mt < 4; ++mt) {
      int g0 = mt * 16, kb0 = k0 + 8 * half;
      v16bf af;
#pragma unroll
      for (int v = 0; v < 8; ++v) {
        int kp = (v < 4) ? 2 * v : 8 + 2 * v;
        af[2 * v]     = (sidx[kb0 + kp]     == g0 + rr) ? one : zero;
        af[2 * v + 1] = (sidx[kb0 + kp + 1] == g0 + rr) ? one : zero;
      }
      acc[mt] = wmma_bf16(af, bf0, acc[mt]);
    }
    if (k0 + 64 < SS) bf0 = load_bt_bf16(vbase, ldv, k0 + 64, d0);
#pragma unroll
    for (int mt = 0; mt < 4; ++mt) {
      int g0 = mt * 16, kb0 = k0 + 32 + 8 * half;
      v16bf af;
#pragma unroll
      for (int v = 0; v < 8; ++v) {
        int kp = (v < 4) ? 2 * v : 8 + 2 * v;
        af[2 * v]     = (sidx[kb0 + kp]     == g0 + rr) ? one : zero;
        af[2 * v + 1] = (sidx[kb0 + kp + 1] == g0 + rr) ? one : zero;
      }
      acc[mt] = wmma_bf16(af, bf1, acc[mt]);
    }
  }

#pragma unroll
  for (int mt = 0; mt < 4; ++mt)
#pragma unroll
    for (int r = 0; r < 8; ++r) {
      int g = mt * 16 + r + 8 * half;
      float norm = 1.0f / ((float)cnt[b * GG + g] + 1.0f);  // sum_s(one_hot) + ASSIGN_EPS
      out_pre[((size_t)b * GG + g) * DD + d0 + rr] = (bf16)(acc[mt][r] * norm);
    }
}

// ============================== host launcher ===============================
extern "C" void kernel_launch(void* const* d_in, const int* in_sizes, int n_in,
                              void* d_out, int out_size, void* d_ws, size_t ws_size,
                              hipStream_t stream) {
  const float* query = (const float*)d_in[0];
  const float* key   = (const float*)d_in[1];
  const float* q_w   = (const float*)d_in[2];
  const float* q_b   = (const float*)d_in[3];
  const float* k_w   = (const float*)d_in[4];
  const float* k_b   = (const float*)d_in[5];
  const float* v_w   = (const float*)d_in[6];
  const float* v_b   = (const float*)d_in[7];
  const float* p_w   = (const float*)d_in[8];
  const float* p_b   = (const float*)d_in[9];

  float* out    = (float*)d_out;                       // [B,G,D] fp32
  float* y_soft = out + (size_t)BB * GG * DD;          // [B,G,S] fp32

  // ---- workspace layout ----
  const size_t WE = (size_t)DD * DD;                   // 589824
  bf16* wsb   = (bf16*)d_ws;
  bf16* qw_t  = wsb;                                   // transposed bf16 weights [N,K]
  bf16* kw_t  = wsb + WE;
  bf16* vw_t  = wsb + 2 * WE;
  bf16* pw_t  = wsb + 3 * WE;
  bf16* q_bf  = wsb + 4 * WE;                          // [B*G, D]
  bf16* k_bf  = q_bf + (size_t)BB * GG * DD;           // [B*S, D]
  bf16* v_t   = k_bf + (size_t)BB * SS * DD;           // [D, B*S] (transposed v)
  bf16* op_bf = v_t + (size_t)BB * SS * DD;            // [B*G, D] (attn @ v)
  int*      idx = (int*)(op_bf + (size_t)BB * GG * DD);
  unsigned* cnt = (unsigned*)(idx + (size_t)BB * SS);

  hipMemsetAsync(cnt, 0, (size_t)BB * GG * sizeof(unsigned), stream);

  // 0) weights -> transposed bf16
  cvt_w_kernel<<<(unsigned)((WE + 255) / 256), 256, 0, stream>>>(q_w, k_w, v_w, p_w, wsb);

  // 1) projections. One block spans all of N=768 -> A read exactly once.
  gemm_kernel<true, false, false><<<dim3(BB * GG / 32, 1), 384, 0, stream>>>(
      query, qw_t, q_b, q_bf, BB * GG, DD, DD, (size_t)DD);
  gemm_kernel<true, false, false><<<dim3(BB * SS / 32, 1), 384, 0, stream>>>(
      key, kw_t, k_b, k_bf, BB * SS, DD, DD, (size_t)DD);
  gemm_kernel<true, true, false><<<dim3(BB * SS / 32, 1), 384, 0, stream>>>(
      key, vw_t, v_b, v_t, BB * SS, DD, DD, (size_t)BB * SS);  // store transposed

  // 2) logits + group-axis softmax + argmax/count; writes y_soft into d_out
  logits_softmax_kernel<<<dim3(SS / 128, BB), 256, 0, stream>>>(q_bf, k_bf, y_soft, idx, cnt);

  // 3) hard-assignment aggregation: out_pre = norm(one_hot) @ v  (WMMA)
  assign_v_kernel<<<dim3(DD / 128, BB), 256, 0, stream>>>(idx, cnt, v_t, op_bf);

  // 4) output projection (bf16 A, fp32 out)
  gemm_kernel<false, false, true><<<dim3(BB * GG / 32, 1), 384, 0, stream>>>(
      op_bf, pw_t, p_b, out, BB * GG, DD, DD, (size_t)DD);

  (void)in_sizes; (void)n_in; (void)out_size; (void)ws_size;
}
